// Aggregator_45981919871430
// MI455X (gfx1250) — matrix operand order repaired and checked by reading, not compiled, not run on hardware
//
#include <hip/hip_runtime.h>
#include <hip/hip_bf16.h>
#include <math.h>

// ---------------------------------------------------------------------------
// Problem constants (from reference)
// ---------------------------------------------------------------------------
#define N_NODES 8001
#define NPAD    8064          // multiple of 64 (flash q-block) and 32 (k-block)
#define E_EDGES 160000
#define IN_DIM  512
#define INTER   128
#define OUT_DIM 64
#define MID     1600
#define BOT     6400

// output layout offsets (flattened tuple, reference return order)
#define OFF_GCN   0
#define OFF_OTW   4
#define OFF_OMID  8
#define OFF_OBOT  12
#define OFF_GF    16
#define OFF_TW    80
#define OFF_MAVG  144
#define OFF_BW    208
#define OFF_WV    272
#define OFF_MF    6672
#define OFF_TB    109072

typedef __attribute__((ext_vector_type(16))) __bf16 v16bf;
typedef __attribute__((ext_vector_type(8)))  float  v8f;

__device__ __forceinline__ v8f wmma_bf16(const v16bf& a, const v16bf& b, const v8f& c) {
    return __builtin_amdgcn_wmma_f32_16x16x32_bf16(false, a, false, b, (short)0, c, false, false);
}

// ---------------------------------------------------------------------------
// CDNA5 async global->LDS copy (GLOBAL_LOAD_ASYNC_TO_LDS_B128, ASYNCcnt).
// No probe-confirmed builtin for this op, so use inline asm per the ISA
// (portable across ROCm 7.2 / amdgpu-toolchain). LDS byte address is the low
// 32 bits of the generic pointer (flat->LDS mapping truncates to addr[31:0]).
// ---------------------------------------------------------------------------
#define USE_ASYNC_LDS 1

__device__ __forceinline__ void async_load_b128_to_lds(const void* gaddr, void* lds) {
#if USE_ASYNC_LDS
    unsigned l = (unsigned)(uintptr_t)lds;
    asm volatile("global_load_async_to_lds_b128 %0, %1, off"
                 :: "v"(l), "v"(gaddr) : "memory");
#else
    const unsigned long long* g = (const unsigned long long*)gaddr;
    unsigned long long* d = (unsigned long long*)lds;
    d[0] = g[0]; d[1] = g[1];
#endif
}

__device__ __forceinline__ void wait_async0() {
#if USE_ASYNC_LDS
    asm volatile("s_wait_asynccnt 0x0" ::: "memory");
#endif
}

// ---------------------------------------------------------------------------
// fp32 -> bf16 conversion helpers
// ---------------------------------------------------------------------------
__global__ void convert_x_kernel(const float* __restrict__ x, __bf16* __restrict__ xbf) {
    int idx = blockIdx.x * blockDim.x + threadIdx.x;
    if (idx >= NPAD * IN_DIM) return;
    xbf[idx] = (idx < N_NODES * IN_DIM) ? (__bf16)x[idx] : (__bf16)0.0f;
}

// W[K x Nout] fp32 row-major  ->  Wt[Nout x K] bf16 (N-major, K contiguous)
__global__ void transpose_w_kernel(const float* __restrict__ W, __bf16* __restrict__ Wt,
                                   int K, int Nout) {
    int idx = blockIdx.x * blockDim.x + threadIdx.x;
    if (idx >= K * Nout) return;
    int k = idx / Nout, n = idx % Nout;
    Wt[(size_t)n * K + k] = (__bf16)W[idx];
}

// ---------------------------------------------------------------------------
// Generic bf16 WMMA GEMM: C[M x Nout] = A[M x K] @ W[K x Nout] (+bias)(ReLU)
// A bf16 row-major, Wt bf16 [Nout x K] (pre-transposed). One wave per 16x16
// output tile; block = 4 waves = 4 n-tiles; grid = (M/16, Nout/64).
// M is NPAD (pad rows of A are zero -> outputs deterministic & finite).
// ---------------------------------------------------------------------------
template<int RELU, int STORE_BF, int STORE_F32>
__global__ __launch_bounds__(128)
void gemm_bf16_kernel(const __bf16* __restrict__ A, const __bf16* __restrict__ Wt,
                      const float* __restrict__ bias,
                      __bf16* __restrict__ outb, float* __restrict__ outf,
                      int K, int Nout) {
    const int lane  = threadIdx.x & 31;
    const int wave  = threadIdx.x >> 5;
    const int lo    = lane & 15;
    const int hi    = lane >> 4;
    const int mtile = blockIdx.x;
    const int ntile = blockIdx.y * 4 + wave;

    const __bf16* arow = A  + (size_t)(mtile * 16 + lo) * K;
    const __bf16* brow = Wt + (size_t)(ntile * 16 + lo) * K;

    v8f acc = {};
    for (int kc = 0; kc < K; kc += 32) {
        v16bf a, b;
#pragma unroll
        for (int i = 0; i < 8; ++i) {
            a[i]     = arow[kc + hi * 16 + i];         // A: K = kc + hi*16 + i
            a[i + 8] = arow[kc + hi * 16 + 8 + i];     //    K = kc + hi*16 + 8 + i
        }
#pragma unroll
        for (int i = 0; i < 16; ++i)
            b[i] = brow[kc + hi * 16 + i];             // B: K = kc + hi*16 + i
        acc = wmma_bf16(a, b, acc);
    }
#pragma unroll
    for (int r = 0; r < 8; ++r) {
        int m = mtile * 16 + r + hi * 8;
        int n = ntile * 16 + lo;
        float v = acc[r] + (bias ? bias[n] : 0.0f);
        if (RELU) v = fmaxf(v, 0.0f);
        if (STORE_BF)  outb[(size_t)m * Nout + n] = (__bf16)v;
        if (STORE_F32) outf[(size_t)m * Nout + n] = v;
    }
}

// ---------------------------------------------------------------------------
// Flash attention: O = softmax(Q K^T) V, streaming over 32-row K/V blocks.
// Q,K,V bf16 [NPAD x D] row-major; O fp32 [NPAD x D]. Keys >= nvalid masked.
// 128 threads = 4 waves; each wave owns 16 query rows (block covers 64 rows).
// K block staged row-major in LDS via async global->LDS DMA (ASYNCcnt);
// V staged transposed through registers (P.V B-frag then contiguous).
// P converted C-layout->A-layout through per-wave LDS scratch (same-wave DS
// ops are in-order).
// ---------------------------------------------------------------------------
template<int D>
__global__ __launch_bounds__(128)
void flash_attn_kernel(const __bf16* __restrict__ Qg, const __bf16* __restrict__ Kg,
                       const __bf16* __restrict__ Vg, float* __restrict__ Og,
                       int nvalid) {
    constexpr int NK = D / 32;   // K chunks for QK^T
    constexpr int NT = D / 16;   // output column tiles
    __shared__ __align__(16) __bf16 Kb[32 * D];
    __shared__ __align__(16) __bf16 Vt[D * 32];
    __shared__ __align__(16) __bf16 Pls[4][16 * 32];

    const int tid   = threadIdx.x;
    const int lane  = tid & 31;
    const int wave  = tid >> 5;
    const int lo    = lane & 15;
    const int hi    = lane >> 4;
    const int qbase = blockIdx.x * 64 + wave * 16;

    // resident Q fragments
    v16bf qa[NK];
    {
        const __bf16* qrow = Qg + (size_t)(qbase + lo) * D;
#pragma unroll
        for (int c = 0; c < NK; ++c)
#pragma unroll
            for (int i = 0; i < 8; ++i) {
                qa[c][i]     = qrow[c * 32 + hi * 16 + i];
                qa[c][i + 8] = qrow[c * 32 + hi * 16 + 8 + i];
            }
    }

    v8f o[NT];
#pragma unroll
    for (int t = 0; t < NT; ++t) o[t] = (v8f){};
    float mrow[8], lrow[8];
#pragma unroll
    for (int r = 0; r < 8; ++r) { mrow[r] = -1e30f; lrow[r] = 0.0f; }

    const int nkb = NPAD / 32;
    for (int kb = 0; kb < nkb; ++kb) {
        // stage K block via async DMA (16B chunks), V block transposed manually
        for (int idx = tid * 8; idx < 32 * D; idx += 128 * 8) {
            int row = idx / D, col = idx % D;
            const __bf16* gk = Kg + (size_t)(kb * 32 + row) * D + col;
            const __bf16* gv = Vg + (size_t)(kb * 32 + row) * D + col;
            async_load_b128_to_lds(gk, &Kb[row * D + col]);
#pragma unroll
            for (int j = 0; j < 8; ++j) Vt[(col + j) * 32 + row] = gv[j];
        }
        if (kb + 1 < nkb)
            __builtin_prefetch(Kg + (size_t)((kb + 1) * 32) * D, 0, 1);
        wait_async0();          // this wave's async copies into Kb complete
        __syncthreads();        // all waves' Kb/Vt writes visible

        // S = Q K^T : two 16x16 tiles (keys 0-15, 16-31)
        v8f s0 = {}, s1 = {};
#pragma unroll
        for (int c = 0; c < NK; ++c) {
            v16bf b0, b1;
#pragma unroll
            for (int i = 0; i < 16; ++i) {
                b0[i] = Kb[(size_t)lo        * D + c * 32 + hi * 16 + i];
                b1[i] = Kb[(size_t)(16 + lo) * D + c * 32 + hi * 16 + i];
            }
            s0 = wmma_bf16(qa[c], b0, s0);
            s1 = wmma_bf16(qa[c], b1, s1);
        }
        // mask padded keys
        if (kb * 32 + lo >= nvalid) {
#pragma unroll
            for (int r = 0; r < 8; ++r) s0[r] = -1e30f;
        }
        if (kb * 32 + 16 + lo >= nvalid) {
#pragma unroll
            for (int r = 0; r < 8; ++r) s1[r] = -1e30f;
        }

        // streaming softmax update (rows r+8*hi; stats across 16-lane groups)
        __bf16* Pw = Pls[wave];
#pragma unroll
        for (int r = 0; r < 8; ++r) {
            float mx = fmaxf(s0[r], s1[r]);
#pragma unroll
            for (int off = 1; off < 16; off <<= 1)
                mx = fmaxf(mx, __shfl_xor(mx, off, 32));
            float mnew  = fmaxf(mrow[r], mx);
            float scale = __expf(mrow[r] - mnew);
            float p0 = __expf(s0[r] - mnew);
            float p1 = __expf(s1[r] - mnew);
            float rs = p0 + p1;
#pragma unroll
            for (int off = 1; off < 16; off <<= 1)
                rs += __shfl_xor(rs, off, 32);
            lrow[r] = lrow[r] * scale + rs;
            mrow[r] = mnew;
#pragma unroll
            for (int t = 0; t < NT; ++t) o[t][r] *= scale;
            Pw[(r + 8 * hi) * 32 + lo]      = (__bf16)p0;
            Pw[(r + 8 * hi) * 32 + 16 + lo] = (__bf16)p1;
        }

        // reload P as A-fragment (same-wave LDS RAW is in-order)
        v16bf pa;
#pragma unroll
        for (int i = 0; i < 8; ++i) {
            pa[i]     = Pw[lo * 32 + hi * 16 + i];
            pa[i + 8] = Pw[lo * 32 + hi * 16 + 8 + i];
        }
        // O += P V
#pragma unroll
        for (int t = 0; t < NT; ++t) {
            v16bf b;
#pragma unroll
            for (int i = 0; i < 16; ++i)
                b[i] = Vt[(t * 16 + lo) * 32 + hi * 16 + i];
            o[t] = wmma_bf16(pa, b, o[t]);
        }
        __syncthreads();
    }

    // epilogue: O /= l
#pragma unroll
    for (int t = 0; t < NT; ++t)
#pragma unroll
        for (int r = 0; r < 8; ++r) {
            size_t row = qbase + r + 8 * hi;
            Og[row * D + t * 16 + lo] = o[t][r] / lrow[r];
        }
}

// ---------------------------------------------------------------------------
// GCN pieces
// ---------------------------------------------------------------------------
__global__ void deg_kernel(const int* __restrict__ dst, float* __restrict__ deg, int E) {
    int e = blockIdx.x * blockDim.x + threadIdx.x;
    if (e < E) atomicAdd(&deg[dst[e]], 1.0f);
}

__global__ void dinv_kernel(const float* __restrict__ deg, float* __restrict__ dinv) {
    int i = blockIdx.x * blockDim.x + threadIdx.x;
    if (i < NPAD) dinv[i] = (i < N_NODES) ? rsqrtf(deg[i] + 1.0f) : 0.0f; // +1 self loop
}

template<int C>
__global__ void gcn_scatter_kernel(const int* __restrict__ src, const int* __restrict__ dst,
                                   const float* __restrict__ dinv,
                                   const float* __restrict__ h, float* __restrict__ out, int E) {
    int e = blockIdx.x * blockDim.x + threadIdx.x;
    if (e >= E) return;
    int s = src[e], d = dst[e];
    float nrm = dinv[s] * dinv[d];
    const float* hr = h + (size_t)s * C;
    float* orow = out + (size_t)d * C;
    for (int c = 0; c < C; ++c) atomicAdd(&orow[c], hr[c] * nrm);
}

// cat = [relu(g_scat + h*dinv^2 + bg) | relu(r)]  as bf16 [NPAD x 2C]
template<int C>
__global__ void combine_kernel(const float* __restrict__ g, const float* __restrict__ h,
                               const float* __restrict__ r, const float* __restrict__ dinv,
                               const float* __restrict__ bg, __bf16* __restrict__ cat) {
    int idx = blockIdx.x * blockDim.x + threadIdx.x;
    if (idx >= NPAD * C) return;
    int i = idx / C, c = idx % C;
    float gv = 0.0f, rv = 0.0f;
    if (i < N_NODES) {
        float di = dinv[i];
        gv = g[idx] + h[idx] * di * di + bg[c];
        rv = r[idx];
    }
    cat[(size_t)i * (2 * C) + c]     = (__bf16)fmaxf(gv, 0.0f);
    cat[(size_t)i * (2 * C) + C + c] = (__bf16)fmaxf(rv, 0.0f);
}

// ---------------------------------------------------------------------------
// Tail: conv / weighting / reductions / classifiers
// ---------------------------------------------------------------------------
__device__ __forceinline__ void atomicMaxF(float* addr, float val) {
    unsigned int* ua = (unsigned int*)addr;
    unsigned int old = *ua;
    while (val > __uint_as_float(old)) {
        unsigned int assumed = old;
        old = atomicCAS(ua, assumed, __float_as_uint(val));
        if (old == assumed) break;
    }
}

__global__ void init_scalar_kernel(float* wvmax) { *wvmax = -1e30f; }

// bw[o][t] = sum_{i,k} bf[4t+k][i] * Wconv[o][i][k] + bconv[o]; wvraw[t*4+o]=bw[o][t]
__global__ void tail_conv_kernel(const float* __restrict__ s2, const float* __restrict__ Wconv,
                                 const float* __restrict__ bconv,
                                 float* __restrict__ wvraw, float* __restrict__ wvmax) {
    int t = blockIdx.x * blockDim.x + threadIdx.x;
    if (t >= MID) return;
    float lmax = -1e30f;
    for (int o = 0; o < 4; ++o) {
        float acc = bconv[o];
        for (int k = 0; k < 4; ++k) {
            const float* row = s2 + (size_t)(1 + MID + 4 * t + k) * OUT_DIM;
            const float* wc  = Wconv + o * 256 + k;   // [o][i][k], stride 4 over i
            for (int i = 0; i < OUT_DIM; ++i) acc += row[i] * wc[i * 4];
        }
        wvraw[4 * t + o] = acc;
        lmax = fmaxf(lmax, acc);
    }
    atomicMaxF(wvmax, lmax);
}

__global__ void tail_wv_kernel(const float* __restrict__ wvraw, const float* __restrict__ wvmax,
                               float* __restrict__ wv_out) {
    int t = blockIdx.x * blockDim.x + threadIdx.x;
    if (t >= MID) return;
    float m = *wvmax;
    float v[4], asum = 0.0f;
    for (int o = 0; o < 4; ++o) { v[o] = wvraw[4 * t + o] / m; asum += fabsf(v[o]); }
    for (int o = 0; o < 4; ++o) wv_out[4 * t + o] = v[o] / asum;
}

// tree_bottom[t][c] = sum_k wv[4t+k]*bf[4t+k][c]   (1600 x 64)
__global__ void tail_tb_kernel(const float* __restrict__ s2, const float* __restrict__ wv,
                               float* __restrict__ tb) {
    int t = blockIdx.x, c = threadIdx.x;
    float acc = 0.0f;
    for (int k = 0; k < 4; ++k)
        acc += wv[4 * t + k] * s2[(size_t)(1 + MID + 4 * t + k) * OUT_DIM + c];
    tb[t * OUT_DIM + c] = acc;
}

// blocks 0..63: mid_avg col; blocks 64..127: bottom_weighted col
__global__ void tail_reduce_kernel(const float* __restrict__ s2, const float* __restrict__ wv,
                                   float* __restrict__ out_mid, float* __restrict__ out_bw) {
    __shared__ float sd[256];
    int c = blockIdx.x & 63;
    bool isMid = blockIdx.x < 64;
    float acc = 0.0f;
    if (isMid) {
        for (int i = 1 + threadIdx.x; i <= MID; i += 256) acc += s2[(size_t)i * OUT_DIM + c];
    } else {
        for (int j = threadIdx.x; j < BOT; j += 256)
            acc += wv[j] * s2[(size_t)(1 + MID + j) * OUT_DIM + c];
    }
    sd[threadIdx.x] = acc;
    __syncthreads();
    for (int s = 128; s > 0; s >>= 1) {
        if (threadIdx.x < s) sd[threadIdx.x] += sd[threadIdx.x + s];
        __syncthreads();
    }
    if (threadIdx.x == 0) {
        float v = sd[0] / (float)MID;
        if (isMid) out_mid[c] = v; else out_bw[c] = v;
    }
}

__global__ void tail_final_kernel(float* __restrict__ out,
                                  const float* Wfc1, const float* bfc1,
                                  const float* Wfc2, const float* bfc2,
                                  const float* Wtw,  const float* btw,
                                  const float* Wmid, const float* bmid,
                                  const float* Wbot, const float* bbot) {
    __shared__ float cat[192];
    __shared__ float gf[64];
    int c = threadIdx.x;             // 64 threads
    cat[c]       = out[OFF_TW + c];
    cat[64 + c]  = out[OFF_MAVG + c];
    cat[128 + c] = out[OFF_BW + c];
    __syncthreads();
    float a = bfc1[c];
    for (int j = 0; j < 192; ++j) a += cat[j] * Wfc1[j * 64 + c];
    gf[c] = a;
    out[OFF_GF + c] = a;
    __syncthreads();
    if (c < 4) {
        float g = bfc2[c], t = btw[c], mm = bmid[c], bb = bbot[c];
        for (int j = 0; j < 64; ++j) {
            g  += fmaxf(gf[j], 0.0f) * Wfc2[j * 4 + c];
            t  += cat[j]       * Wtw[j * 4 + c];
            mm += cat[64 + j]  * Wmid[j * 4 + c];
            bb += cat[128 + j] * Wbot[j * 4 + c];
        }
        out[OFF_GCN + c]  = g;
        out[OFF_OTW + c]  = t;
        out[OFF_OMID + c] = mm;
        out[OFF_OBOT + c] = bb;
    }
}

// ---------------------------------------------------------------------------
// Host orchestration
// ---------------------------------------------------------------------------
extern "C" void kernel_launch(void* const* d_in, const int* in_sizes, int n_in,
                              void* d_out, int out_size, void* d_ws, size_t ws_size,
                              hipStream_t stream) {
    const float* x    = (const float*)d_in[0];
    const int*   ei   = (const int*)d_in[1];
    const float* Wq1  = (const float*)d_in[2];  const float* bq1 = (const float*)d_in[3];
    const float* Wk1  = (const float*)d_in[4];  const float* bk1 = (const float*)d_in[5];
    const float* Wv1  = (const float*)d_in[6];  const float* bv1 = (const float*)d_in[7];
    const float* Wq2  = (const float*)d_in[8];  const float* bq2 = (const float*)d_in[9];
    const float* Wk2  = (const float*)d_in[10]; const float* bk2 = (const float*)d_in[11];
    const float* Wv2  = (const float*)d_in[12]; const float* bv2 = (const float*)d_in[13];
    const float* Wg1  = (const float*)d_in[14]; const float* bg1 = (const float*)d_in[15];
    const float* Wg2  = (const float*)d_in[16]; const float* bg2 = (const float*)d_in[17];
    const float* Ws1  = (const float*)d_in[18]; const float* bs1 = (const float*)d_in[19];
    const float* Ws2  = (const float*)d_in[20]; const float* bs2 = (const float*)d_in[21];
    const float* Wfc1 = (const float*)d_in[22]; const float* bfc1 = (const float*)d_in[23];
    const float* Wfc2 = (const float*)d_in[24]; const float* bfc2 = (const float*)d_in[25];
    const float* Wtw  = (const float*)d_in[26]; const float* btw = (const float*)d_in[27];
    const float* Wmid = (const float*)d_in[28]; const float* bmid = (const float*)d_in[29];
    const float* Wbot = (const float*)d_in[30]; const float* bbot = (const float*)d_in[31];
    const float* Wconv = (const float*)d_in[32]; const float* bconv = (const float*)d_in[33];
    float* out = (float*)d_out;

    const int* src = ei;
    const int* dst = ei + E_EDGES;

    // bump allocator over workspace
    char* wptr = (char*)d_ws;
    auto alloc = [&](size_t bytes) -> char* {
        char* p = wptr;
        wptr += (bytes + 255) & ~(size_t)255;
        return p;
    };

    __bf16* xbf   = (__bf16*)alloc((size_t)NPAD * IN_DIM * 2);
    __bf16* Wq1t  = (__bf16*)alloc((size_t)INTER * IN_DIM * 2);
    __bf16* Wk1t  = (__bf16*)alloc((size_t)INTER * IN_DIM * 2);
    __bf16* Wv1t  = (__bf16*)alloc((size_t)INTER * IN_DIM * 2);
    __bf16* Wg1t  = (__bf16*)alloc((size_t)INTER * IN_DIM * 2);
    __bf16* Ws1t  = (__bf16*)alloc((size_t)INTER * (2 * INTER) * 2);
    __bf16* Wq2t  = (__bf16*)alloc((size_t)OUT_DIM * INTER * 2);
    __bf16* Wk2t  = (__bf16*)alloc((size_t)OUT_DIM * INTER * 2);
    __bf16* Wv2t  = (__bf16*)alloc((size_t)OUT_DIM * INTER * 2);
    __bf16* Wg2t  = (__bf16*)alloc((size_t)OUT_DIM * INTER * 2);
    __bf16* Ws2t  = (__bf16*)alloc((size_t)OUT_DIM * (2 * OUT_DIM) * 2);
    __bf16* q1b   = (__bf16*)alloc((size_t)NPAD * INTER * 2);
    __bf16* k1b   = (__bf16*)alloc((size_t)NPAD * INTER * 2);
    __bf16* v1b   = (__bf16*)alloc((size_t)NPAD * INTER * 2);
    float*  h1f   = (float*)alloc((size_t)NPAD * INTER * 4);
    float*  r1f   = (float*)alloc((size_t)NPAD * INTER * 4);
    float*  deg   = (float*)alloc((size_t)NPAD * 4);
    float*  dinv  = (float*)alloc((size_t)NPAD * 4);
    float*  g1    = (float*)alloc((size_t)NPAD * INTER * 4);
    __bf16* cat1  = (__bf16*)alloc((size_t)NPAD * (2 * INTER) * 2);
    __bf16* s1b   = (__bf16*)alloc((size_t)NPAD * INTER * 2);
    __bf16* q2b   = (__bf16*)alloc((size_t)NPAD * OUT_DIM * 2);
    __bf16* k2b   = (__bf16*)alloc((size_t)NPAD * OUT_DIM * 2);
    __bf16* v2b   = (__bf16*)alloc((size_t)NPAD * OUT_DIM * 2);
    float*  h2f   = (float*)alloc((size_t)NPAD * OUT_DIM * 4);
    float*  r2f   = (float*)alloc((size_t)NPAD * OUT_DIM * 4);
    float*  g2    = (float*)alloc((size_t)NPAD * OUT_DIM * 4);
    __bf16* cat2  = (__bf16*)alloc((size_t)NPAD * (2 * OUT_DIM) * 2);
    float*  s2f   = (float*)alloc((size_t)NPAD * OUT_DIM * 4);
    float*  wvraw = (float*)alloc((size_t)BOT * 4);
    float*  wvmax = (float*)alloc(256);

    // zero accumulation buffers (re-done every call; graph-capture safe)
    hipMemsetAsync(deg, 0, (size_t)NPAD * 4, stream);
    hipMemsetAsync(g1, 0, (size_t)NPAD * INTER * 4, stream);
    hipMemsetAsync(g2, 0, (size_t)NPAD * OUT_DIM * 4, stream);
    init_scalar_kernel<<<1, 1, 0, stream>>>(wvmax);

    // conversions
    convert_x_kernel<<<(NPAD * IN_DIM + 255) / 256, 256, 0, stream>>>(x, xbf);
    auto tw_launch = [&](const float* W, __bf16* Wt, int K, int Nout) {
        transpose_w_kernel<<<(K * Nout + 255) / 256, 256, 0, stream>>>(W, Wt, K, Nout);
    };
    tw_launch(Wq1, Wq1t, IN_DIM, INTER);
    tw_launch(Wk1, Wk1t, IN_DIM, INTER);
    tw_launch(Wv1, Wv1t, IN_DIM, INTER);
    tw_launch(Wg1, Wg1t, IN_DIM, INTER);
    tw_launch(Ws1, Ws1t, 2 * INTER, INTER);
    tw_launch(Wq2, Wq2t, INTER, OUT_DIM);
    tw_launch(Wk2, Wk2t, INTER, OUT_DIM);
    tw_launch(Wv2, Wv2t, INTER, OUT_DIM);
    tw_launch(Wg2, Wg2t, INTER, OUT_DIM);
    tw_launch(Ws2, Ws2t, 2 * OUT_DIM, OUT_DIM);

    const dim3 g128(NPAD / 16, INTER / 64);   // Nout=128
    const dim3 g64(NPAD / 16, 1);             // Nout=64

    // layer 1 projections
    gemm_bf16_kernel<0, 1, 0><<<g128, 128, 0, stream>>>(xbf, Wq1t, bq1, q1b, nullptr, IN_DIM, INTER);
    gemm_bf16_kernel<0, 1, 0><<<g128, 128, 0, stream>>>(xbf, Wk1t, bk1, k1b, nullptr, IN_DIM, INTER);
    gemm_bf16_kernel<0, 1, 0><<<g128, 128, 0, stream>>>(xbf, Wv1t, bv1, v1b, nullptr, IN_DIM, INTER);
    gemm_bf16_kernel<0, 0, 1><<<g128, 128, 0, stream>>>(xbf, Wg1t, nullptr, nullptr, h1f, IN_DIM, INTER);

    // attention 1 + gcn 1
    flash_attn_kernel<INTER><<<NPAD / 64, 128, 0, stream>>>(q1b, k1b, v1b, r1f, N_NODES);
    deg_kernel<<<(E_EDGES + 255) / 256, 256, 0, stream>>>(dst, deg, E_EDGES);
    dinv_kernel<<<(NPAD + 255) / 256, 256, 0, stream>>>(deg, dinv);
    gcn_scatter_kernel<INTER><<<(E_EDGES + 255) / 256, 256, 0, stream>>>(src, dst, dinv, h1f, g1, E_EDGES);
    combine_kernel<INTER><<<(NPAD * INTER + 255) / 256, 256, 0, stream>>>(g1, h1f, r1f, dinv, bg1, cat1);

    // s1 = relu(cat1 @ Ws1 + bs1)
    gemm_bf16_kernel<1, 1, 0><<<g128, 128, 0, stream>>>(cat1, Ws1t, bs1, s1b, nullptr, 2 * INTER, INTER);

    // layer 2 projections
    gemm_bf16_kernel<0, 1, 0><<<g64, 128, 0, stream>>>(s1b, Wq2t, bq2, q2b, nullptr, INTER, OUT_DIM);
    gemm_bf16_kernel<0, 1, 0><<<g64, 128, 0, stream>>>(s1b, Wk2t, bk2, k2b, nullptr, INTER, OUT_DIM);
    gemm_bf16_kernel<0, 1, 0><<<g64, 128, 0, stream>>>(s1b, Wv2t, bv2, v2b, nullptr, INTER, OUT_DIM);
    gemm_bf16_kernel<0, 0, 1><<<g64, 128, 0, stream>>>(s1b, Wg2t, nullptr, nullptr, h2f, INTER, OUT_DIM);

    // attention 2 + gcn 2
    flash_attn_kernel<OUT_DIM><<<NPAD / 64, 128, 0, stream>>>(q2b, k2b, v2b, r2f, N_NODES);
    gcn_scatter_kernel<OUT_DIM><<<(E_EDGES + 255) / 256, 256, 0, stream>>>(src, dst, dinv, h2f, g2, E_EDGES);
    combine_kernel<OUT_DIM><<<(NPAD * OUT_DIM + 255) / 256, 256, 0, stream>>>(g2, h2f, r2f, dinv, bg2, cat2);

    // s2 = relu(cat2 @ Ws2 + bs2)  (fp32, feeds the tail)
    gemm_bf16_kernel<1, 0, 1><<<g64, 128, 0, stream>>>(cat2, Ws2t, bs2, nullptr, s2f, 2 * OUT_DIM, OUT_DIM);

    // tail
    hipMemcpyAsync(out + OFF_MF, s2f + OUT_DIM, (size_t)MID * OUT_DIM * 4,
                   hipMemcpyDeviceToDevice, stream);                     // mid_features
    hipMemcpyAsync(out + OFF_TW, s2f, (size_t)OUT_DIM * 4,
                   hipMemcpyDeviceToDevice, stream);                     // tw
    tail_conv_kernel<<<(MID + 255) / 256, 256, 0, stream>>>(s2f, Wconv, bconv, wvraw, wvmax);
    tail_wv_kernel<<<(MID + 255) / 256, 256, 0, stream>>>(wvraw, wvmax, out + OFF_WV);
    tail_tb_kernel<<<MID, OUT_DIM, 0, stream>>>(s2f, out + OFF_WV, out + OFF_TB);
    tail_reduce_kernel<<<128, 256, 0, stream>>>(s2f, out + OFF_WV, out + OFF_MAVG, out + OFF_BW);
    tail_final_kernel<<<1, 64, 0, stream>>>(out, Wfc1, bfc1, Wfc2, bfc2,
                                            Wtw, btw, Wmid, bmid, Wbot, bbot);
}
